// SwinTransformerBlock_51917564674157
// MI455X (gfx1250) — matrix operand order repaired and checked by reading, not compiled, not run on hardware
//
#include <hip/hip_runtime.h>
#include <hip/hip_bf16.h>

typedef __attribute__((ext_vector_type(16))) _Float16 v16h;
typedef __attribute__((ext_vector_type(8)))  _Float16 v8h;
typedef __attribute__((ext_vector_type(8)))  float    v8f;

union V16U { v16h v; v8h h[2]; };

#define EMB   192
#define NHD   6
#define HD    32
#define NTOK  64
#define DFF   768
#define LNEPS 1e-7f

// ---- WMMA operand loaders (CDNA5 ISA 7.12.2 layouts, wave32) ----

// A-matrix 16x32 f16: lane L -> M=L&15; K-base=(L>=16)?8:0; halves cover
// K in [kb,kb+8) then [kb+16,kb+24)  (two contiguous 16B chunks)
static __device__ __forceinline__ v16h load_a16x32(const _Float16* base, int stride) {
  const int lane = threadIdx.x & 31;
  const _Float16* p = base + (lane & 15) * stride + ((lane >> 4) << 3);
  V16U u;
  u.h[0] = *(const v8h*)(p);
  u.h[1] = *(const v8h*)(p + 16);
  return u.v;
}

// B-matrix 32x16 f16: lane L -> N=L&15; K-base=(L>=16)?8:0; K in [kb,kb+8) then [kb+16,kb+24)
static __device__ __forceinline__ v16h load_b32x16(const _Float16* base, int stride) {
  const int lane = threadIdx.x & 31;
  const _Float16* p = base + (lane & 15) * stride + ((lane >> 4) << 3);
  V16U u;
  u.h[0] = *(const v8h*)(p);
  u.h[1] = *(const v8h*)(p + 16);
  return u.v;
}

static __device__ __forceinline__ v8f wmma_f16(v16h a, v16h b, v8f c) {
  return __builtin_amdgcn_wmma_f32_16x16x32_f16(false, a, false, b, (short)0, c, false, false);
}

// ---- Prep: transpose + fp32->f16 all weight matrices into workspace ----
// ws layout (halves): wqT[192x192] wkT wvT wpT | w1T[768x192] | w2T[192x768]
__global__ void swin_prep_weights(const float* __restrict__ wq, const float* __restrict__ wk,
                                  const float* __restrict__ wv, const float* __restrict__ wp,
                                  const float* __restrict__ w1, const float* __restrict__ w2,
                                  _Float16* __restrict__ ws) {
  int idx = blockIdx.x * 256 + threadIdx.x;
  if (idx < 147456) {                       // 4 x 192x192, T[n*192+k] = W[k*192+n]
    int mat = idx / 36864, r = idx % 36864, n = r / 192, k = r % 192;
    const float* w = (mat == 0) ? wq : (mat == 1) ? wk : (mat == 2) ? wv : wp;
    ws[idx] = (_Float16)w[k * 192 + n];
  } else if (idx < 294912) {                // w1 (192x768) -> w1T (768 rows x 192)
    int r = idx - 147456, n = r / 192, k = r % 192;
    ws[idx] = (_Float16)w1[k * 768 + n];
  } else if (idx < 442368) {                // w2 (768x192) -> w2T (192 rows x 768)
    int r = idx - 294912, n = r / 768, k = r % 768;
    ws[idx] = (_Float16)w2[k * 192 + n];
  }
}

// ---- Fused Swin block: one workgroup (8 wave32) per 8x8 window ----
__global__ __launch_bounds__(256, 1)
void swin_block_kernel(const float* __restrict__ x, const _Float16* __restrict__ wts,
                       const float* __restrict__ bq, const float* __restrict__ bk,
                       const float* __restrict__ bv, const float* __restrict__ bp,
                       const float* __restrict__ rpb,
                       const float* __restrict__ g1, const float* __restrict__ be1,
                       const float* __restrict__ g2, const float* __restrict__ be2,
                       const float* __restrict__ b1m, const float* __restrict__ b2m,
                       float* __restrict__ out) {
  extern __shared__ char smem[];
  // phased region map (bytes):
  _Float16* sX   = (_Float16*)(smem);           // [0,24K)   xw f16 -> attn-out f16
  _Float16* sQ   = (_Float16*)(smem + 24576);   // [24,48K)  Q*scale f16
  _Float16* sK   = (_Float16*)(smem + 49152);   // [48,72K)  K f16
  _Float16* sVt  = (_Float16*)(smem + 73728);   // [72,96K)  V transposed (dim-major)
  float*    sS   = (float*)(smem + 98304);      // [96,192K) scores f32 (6x64x64)
  _Float16* sP   = (_Float16*)(smem + 24576);   // [24,72K)  probs f16 (over dead Q,K)
  _Float16* sX1h = (_Float16*)(smem + 24576);   // [24,48K)  x1 f16 (over dead probs)
  float*    sO2  = (float*)(smem + 49152);      // [48,96K)  mlp out f32 (over dead K,Vt)
  float*    sOp  = (float*)(smem + 98304);      // [96,144K) proj out f32 (over dead S)
  _Float16* sH   = (_Float16*)(smem + 98304);   // [96,192K) mlp hidden f16
  float*    sRes = (float*)(smem + 196608);     // [192,240K) shortcut -> x1 f32
  float*    sMean= (float*)(smem + 245760);
  float*    sRstd= (float*)(smem + 246016);

  const int tid  = threadIdx.x;
  const int wvid = tid >> 5;
  const int lane = tid & 31;
  const int n    = lane & 15;
  const int mb   = (lane >> 4) * 8;
  const int blk  = blockIdx.x;
  const int b    = blk >> 8;
  const int wid  = blk & 255;
  const int wy   = wid >> 4;
  const int wx   = wid & 15;

  const _Float16* wqT = wts;
  const _Float16* wkT = wts + 36864;
  const _Float16* wvT = wts + 73728;
  const _Float16* wpT = wts + 110592;
  const _Float16* w1T = wts + 147456;
  const _Float16* w2T = wts + 294912;

  // Phase 0: async-copy shifted window tile (fp32, 16B chunks) straight to LDS
  // (GLOBAL_LOAD_ASYNC_TO_LDS_B128, tracked by ASYNCcnt; roll -4 == (idx+4)&127)
  for (int idx = tid; idx < 3072; idx += 256) {
    int t = idx / 48, c4 = idx - t * 48;
    int h = (wy * 8 + (t >> 3) + 4) & 127;
    int w = (wx * 8 + (t & 7) + 4) & 127;
    const float* gp = x + ((size_t)b * 16384 + h * 128 + w) * EMB + c4 * 4;
    unsigned lds = (unsigned)(uintptr_t)(sRes + t * EMB + c4 * 4);
    asm volatile("global_load_async_to_lds_b128 %0, %1, off"
                 :: "v"(lds), "v"(gp) : "memory");
  }
  asm volatile("s_wait_asynccnt 0" ::: "memory");
  __syncthreads();
  for (int idx = tid; idx < NTOK * EMB; idx += 256) sX[idx] = (_Float16)sRes[idx];
  __syncthreads();

  // Phase 1: Q,K,V = xw @ W + b.  Two N-tiles per iteration (independent
  // accumulator chains share the A fragment). 4 mt x 6 pairs = 24 iters each.
  const float scale = 0.17677669529663687f;  // 32^-0.5, folded into Q
  for (int t = wvid; t < 24; t += 8) {       // Q
    int mt = t / 6, np = t % 6, n0 = np * 32, n1 = n0 + 16;
    float b0 = bq[n0 + n], b1 = bq[n1 + n];
    v8f a0, a1; for (int i = 0; i < 8; ++i) { a0[i] = b0; a1[i] = b1; }
#pragma unroll
    for (int kt = 0; kt < 6; ++kt) {
      v16h a = load_a16x32(sX + mt * 16 * EMB + kt * 32, EMB);
      a0 = wmma_f16(a, load_b32x16(wqT + n0 * EMB + kt * 32, EMB), a0);
      a1 = wmma_f16(a, load_b32x16(wqT + n1 * EMB + kt * 32, EMB), a1);
    }
#pragma unroll
    for (int r = 0; r < 8; ++r) {
      sQ[(mt*16+mb+r)*EMB + n0 + n] = (_Float16)(a0[r] * scale);
      sQ[(mt*16+mb+r)*EMB + n1 + n] = (_Float16)(a1[r] * scale);
    }
  }
  for (int t = wvid; t < 24; t += 8) {       // K
    int mt = t / 6, np = t % 6, n0 = np * 32, n1 = n0 + 16;
    float b0 = bk[n0 + n], b1 = bk[n1 + n];
    v8f a0, a1; for (int i = 0; i < 8; ++i) { a0[i] = b0; a1[i] = b1; }
#pragma unroll
    for (int kt = 0; kt < 6; ++kt) {
      v16h a = load_a16x32(sX + mt * 16 * EMB + kt * 32, EMB);
      a0 = wmma_f16(a, load_b32x16(wkT + n0 * EMB + kt * 32, EMB), a0);
      a1 = wmma_f16(a, load_b32x16(wkT + n1 * EMB + kt * 32, EMB), a1);
    }
#pragma unroll
    for (int r = 0; r < 8; ++r) {
      sK[(mt*16+mb+r)*EMB + n0 + n] = (_Float16)a0[r];
      sK[(mt*16+mb+r)*EMB + n1 + n] = (_Float16)a1[r];
    }
  }
  for (int t = wvid; t < 24; t += 8) {       // V (stored transposed, dim-major)
    int mt = t / 6, np = t % 6, n0 = np * 32, n1 = n0 + 16;
    float b0 = bv[n0 + n], b1 = bv[n1 + n];
    v8f a0, a1; for (int i = 0; i < 8; ++i) { a0[i] = b0; a1[i] = b1; }
#pragma unroll
    for (int kt = 0; kt < 6; ++kt) {
      v16h a = load_a16x32(sX + mt * 16 * EMB + kt * 32, EMB);
      a0 = wmma_f16(a, load_b32x16(wvT + n0 * EMB + kt * 32, EMB), a0);
      a1 = wmma_f16(a, load_b32x16(wvT + n1 * EMB + kt * 32, EMB), a1);
    }
#pragma unroll
    for (int r = 0; r < 8; ++r) {            // lane-contiguous -> ds_store_b128
      sVt[(n0+n)*64 + mt*16+mb+r] = (_Float16)a0[r];
      sVt[(n1+n)*64 + mt*16+mb+r] = (_Float16)a1[r];
    }
  }
  __syncthreads();

  // Phase 2: S = (Q*scale) @ K^T  (hd=32 == one WMMA K-step; 2 N-tiles/iter)
  for (int t = wvid; t < 48; t += 8) {
    int head = t / 8, r4 = t % 8, mt = r4 >> 1, np = r4 & 1;
    int n0 = np * 32, n1 = n0 + 16;
    v8f a0, a1; for (int i = 0; i < 8; ++i) { a0[i] = 0.f; a1[i] = 0.f; }
    v16h a = load_a16x32(sQ + mt * 16 * EMB + head * HD, EMB);
    a0 = wmma_f16(a, load_b32x16(sK + n0 * EMB + head * HD, EMB), a0);
    a1 = wmma_f16(a, load_b32x16(sK + n1 * EMB + head * HD, EMB), a1);
#pragma unroll
    for (int r = 0; r < 8; ++r) {
      sS[head*4096 + (mt*16+mb+r)*64 + n0 + n] = a0[r];
      sS[head*4096 + (mt*16+mb+r)*64 + n1 + n] = a1[r];
    }
  }
  __syncthreads();

  // Phase 3: + rel-pos bias + shift mask (inline) -> softmax -> f16 probs
  for (int r = tid; r < NHD * 64; r += 256) {
    int head = r / 64, i = r - head * 64;
    int tyi = i >> 3, txi = i & 7;
    int hs = wy * 8 + tyi, wc = wx * 8 + txi;
    int ci = ((hs < 120) ? 0 : (hs < 124) ? 1 : 2) * 3 + ((wc < 120) ? 0 : (wc < 124) ? 1 : 2);
    float* row = sS + head * 4096 + i * 64;
    float mx = -1e30f;
    for (int j = 0; j < 64; ++j) {
      int tyj = j >> 3, txj = j & 7;
      int hj = wy * 8 + tyj, wj = wx * 8 + txj;
      int cj = ((hj < 120) ? 0 : (hj < 124) ? 1 : 2) * 3 + ((wj < 120) ? 0 : (wj < 124) ? 1 : 2);
      float v = row[j] + rpb[((tyi - tyj + 7) * 15 + (txi - txj + 7)) * NHD + head];
      if (ci != cj) v -= 100.f;
      row[j] = v;
      mx = fmaxf(mx, v);
    }
    float sum = 0.f;
    for (int j = 0; j < 64; ++j) { float e = expf(row[j] - mx); sum += e; row[j] = e; }
    float inv = 1.f / sum;
    _Float16* prow = sP + head * 4096 + i * 64;
    for (int j = 0; j < 64; ++j) prow[j] = (_Float16)(row[j] * inv);
  }
  __syncthreads();

  // Phase 4: O = P @ V  (K=64 -> 2 steps; both head-dim N-tiles per iter)
  for (int t = wvid; t < 24; t += 8) {
    int head = t / 4, mt = t % 4;
    v8f a0, a1; for (int i = 0; i < 8; ++i) { a0[i] = 0.f; a1[i] = 0.f; }
#pragma unroll
    for (int kt = 0; kt < 2; ++kt) {
      v16h a = load_a16x32(sP + head * 4096 + mt * 16 * 64 + kt * 32, 64);
      a0 = wmma_f16(a, load_b32x16(sVt + (head * HD)      * 64 + kt * 32, 64), a0);
      a1 = wmma_f16(a, load_b32x16(sVt + (head * HD + 16) * 64 + kt * 32, 64), a1);
    }
#pragma unroll
    for (int r = 0; r < 8; ++r) {
      sX[(mt*16+mb+r)*EMB + head*HD      + n] = (_Float16)a0[r];
      sX[(mt*16+mb+r)*EMB + head*HD + 16 + n] = (_Float16)a1[r];
    }
  }
  __syncthreads();

  // Phase 5: proj = O @ wp + bp  (f32 into sOp)
  for (int t = wvid; t < 24; t += 8) {
    int mt = t / 6, np = t % 6, n0 = np * 32, n1 = n0 + 16;
    float b0 = bp[n0 + n], b1 = bp[n1 + n];
    v8f a0, a1; for (int i = 0; i < 8; ++i) { a0[i] = b0; a1[i] = b1; }
#pragma unroll
    for (int kt = 0; kt < 6; ++kt) {
      v16h a = load_a16x32(sX + mt * 16 * EMB + kt * 32, EMB);
      a0 = wmma_f16(a, load_b32x16(wpT + n0 * EMB + kt * 32, EMB), a0);
      a1 = wmma_f16(a, load_b32x16(wpT + n1 * EMB + kt * 32, EMB), a1);
    }
#pragma unroll
    for (int r = 0; r < 8; ++r) {
      sOp[(mt*16+mb+r)*EMB + n0 + n] = a0[r];
      sOp[(mt*16+mb+r)*EMB + n1 + n] = a1[r];
    }
  }
  __syncthreads();

  // Phase 6: x1 = LN(shortcut + proj) -> sRes (f32) and sX1h (f16)
  for (int idx = tid; idx < NTOK * EMB; idx += 256) sRes[idx] += sOp[idx];
  __syncthreads();
  if (tid < 64) {
    const float* row = sRes + tid * EMB;
    float s = 0.f; for (int c = 0; c < EMB; ++c) s += row[c];
    float m = s / EMB;
    float v = 0.f; for (int c = 0; c < EMB; ++c) { float d = row[c] - m; v += d * d; }
    sMean[tid] = m; sRstd[tid] = rsqrtf(v / EMB + LNEPS);
  }
  __syncthreads();
  for (int idx = tid; idx < NTOK * EMB; idx += 256) {
    int t = idx / EMB, c = idx - t * EMB;
    float x1 = (sRes[idx] - sMean[t]) * sRstd[t] * g1[c] + be1[c];
    sRes[idx] = x1; sX1h[idx] = (_Float16)x1;
  }
  __syncthreads();

  // Phase 7: hidden = GELU(x1 @ w1 + b1)  (4 mt x 24 pairs) -> f16 in sH
  for (int t = wvid; t < 96; t += 8) {
    int mt = t / 24, np = t % 24, n0 = np * 32, n1 = n0 + 16;
    float b0 = b1m[n0 + n], b1 = b1m[n1 + n];
    v8f a0, a1; for (int i = 0; i < 8; ++i) { a0[i] = b0; a1[i] = b1; }
#pragma unroll
    for (int kt = 0; kt < 6; ++kt) {
      v16h a = load_a16x32(sX1h + mt * 16 * EMB + kt * 32, EMB);
      a0 = wmma_f16(a, load_b32x16(w1T + n0 * EMB + kt * 32, EMB), a0);
      a1 = wmma_f16(a, load_b32x16(w1T + n1 * EMB + kt * 32, EMB), a1);
    }
#pragma unroll
    for (int r = 0; r < 8; ++r) {
      float x0 = a0[r], x1 = a1[r];
      sH[(mt*16+mb+r)*DFF + n0 + n] = (_Float16)(0.5f*x0*(1.f+erff(x0*0.7071067811865476f)));
      sH[(mt*16+mb+r)*DFF + n1 + n] = (_Float16)(0.5f*x1*(1.f+erff(x1*0.7071067811865476f)));
    }
  }
  __syncthreads();

  // Phase 8: mlp = hidden @ w2 + b2  (K=768 -> 24 steps; 2 N-tiles/iter)
  for (int t = wvid; t < 24; t += 8) {
    int mt = t / 6, np = t % 6, n0 = np * 32, n1 = n0 + 16;
    float b0 = b2m[n0 + n], b1 = b2m[n1 + n];
    v8f a0, a1; for (int i = 0; i < 8; ++i) { a0[i] = b0; a1[i] = b1; }
    for (int kt = 0; kt < 24; ++kt) {
      v16h a = load_a16x32(sH + mt * 16 * DFF + kt * 32, DFF);
      a0 = wmma_f16(a, load_b32x16(w2T + n0 * DFF + kt * 32, DFF), a0);
      a1 = wmma_f16(a, load_b32x16(w2T + n1 * DFF + kt * 32, DFF), a1);
    }
#pragma unroll
    for (int r = 0; r < 8; ++r) {
      sO2[(mt*16+mb+r)*EMB + n0 + n] = a0[r];
      sO2[(mt*16+mb+r)*EMB + n1 + n] = a1[r];
    }
  }
  __syncthreads();

  // Phase 9: out = LN(x1 + mlp), scatter with unshift (same (+4)&127 mapping)
  for (int idx = tid; idx < NTOK * EMB; idx += 256) sO2[idx] += sRes[idx];
  __syncthreads();
  if (tid < 64) {
    const float* row = sO2 + tid * EMB;
    float s = 0.f; for (int c = 0; c < EMB; ++c) s += row[c];
    float m = s / EMB;
    float v = 0.f; for (int c = 0; c < EMB; ++c) { float d = row[c] - m; v += d * d; }
    sMean[tid] = m; sRstd[tid] = rsqrtf(v / EMB + LNEPS);
  }
  __syncthreads();
  for (int idx = tid; idx < NTOK * EMB; idx += 256) {
    int t = idx / EMB, c = idx - t * EMB;
    int h = (wy * 8 + (t >> 3) + 4) & 127;
    int w = (wx * 8 + (t & 7) + 4) & 127;
    out[((size_t)b * 16384 + h * 128 + w) * EMB + c] =
        (sO2[idx] - sMean[t]) * sRstd[t] * g2[c] + be2[c];
  }
}

extern "C" void kernel_launch(void* const* d_in, const int* in_sizes, int n_in,
                              void* d_out, int out_size, void* d_ws, size_t ws_size,
                              hipStream_t stream) {
  const float* x   = (const float*)d_in[0];
  const float* wq  = (const float*)d_in[1];  const float* bq  = (const float*)d_in[2];
  const float* wk  = (const float*)d_in[3];  const float* bk  = (const float*)d_in[4];
  const float* wv  = (const float*)d_in[5];  const float* bv  = (const float*)d_in[6];
  const float* wp  = (const float*)d_in[7];  const float* bp  = (const float*)d_in[8];
  const float* rpb = (const float*)d_in[9];
  const float* g1  = (const float*)d_in[10]; const float* be1 = (const float*)d_in[11];
  const float* g2  = (const float*)d_in[12]; const float* be2 = (const float*)d_in[13];
  const float* w1  = (const float*)d_in[14]; const float* b1m = (const float*)d_in[15];
  const float* w2  = (const float*)d_in[16]; const float* b2m = (const float*)d_in[17];
  float* out = (float*)d_out;
  _Float16* wts = (_Float16*)d_ws;  // needs 884,736 bytes of d_ws

  (void)in_sizes; (void)n_in; (void)out_size; (void)ws_size;

  const size_t smem_bytes = 246272;  // 240.5 KB dynamic LDS (gfx1250: 320 KB/WGP)
  (void)hipFuncSetAttribute((const void*)swin_block_kernel,
                            hipFuncAttributeMaxDynamicSharedMemorySize, (int)smem_bytes);

  swin_prep_weights<<<1728, 256, 0, stream>>>(wq, wk, wv, wp, w1, w2, wts);
  swin_block_kernel<<<2048, 256, smem_bytes, stream>>>(
      x, wts, bq, bk, bv, bp, rpb, g1, be1, g2, be2, b1m, b2m, out);
}
